// HolographicMixer_532575944913
// MI455X (gfx1250) — matrix operand order repaired and checked by reading, not compiled, not run on hardware
//
#include <hip/hip_runtime.h>
#include <cstdint>
#include <cstddef>

// ---------------------------------------------------------------------------
// HolographicMixer on gfx1250:
//   out = x + gate * (cumsum_S(x) @ Circulant(k_eff))
//   k_eff = circ_corr(bind_key, unbind_key)
// Split-bf16 (hi/lo) GEMM on v_wmma_f32_16x16x32_bf16 for fp32-grade accuracy,
// with double-buffered GLOBAL_LOAD_ASYNC_TO_LDS staging (ASYNCcnt pipeline).
// ---------------------------------------------------------------------------

typedef __bf16 bf16_t;
typedef __attribute__((ext_vector_type(16))) __bf16 v16bf;
typedef __attribute__((ext_vector_type(8)))  float  v8f;

#define BATCH 8
#define SEQ   4096
#define DIM   1024
#define NCHUNK 16
#define CHUNK  256     // SEQ / NCHUNK

// GEMM tiling
#define TS  128        // S-tile rows per block
#define TD  128        // D-tile cols per block
#define TK  32         // K step (one bf16 WMMA K)
#define LDK 40         // padded LDS row stride (elements); 40*2B = 80B, 16B aligned
#define ARR_B (TS * LDK * 2)   // bytes per LDS tile array (10240)

union FragBF { v16bf v; uint4 q[2]; };

__device__ __forceinline__ v8f wmma_bf16(v16bf a, v16bf b, v8f c) {
    // (neg_a, A, neg_b, B, c_mod, C, reuse_a, reuse_b)
    return __builtin_amdgcn_wmma_f32_16x16x32_bf16(false, a, false, b, (short)0, c,
                                                   false, false);
}

// async 16B global -> LDS copy (per active lane); tracked by ASYNCcnt
__device__ __forceinline__ void async_copy16(uint32_t lds_off, const void* gptr) {
    asm volatile("global_load_async_to_lds_b128 %0, %1, off"
                 :: "v"(lds_off), "v"((uint64_t)(uintptr_t)gptr)
                 : "memory");
}
__device__ __forceinline__ void wait_async8() {
    asm volatile("s_wait_asynccnt 0x8" ::: "memory");
}
__device__ __forceinline__ void wait_async0() {
    asm volatile("s_wait_asynccnt 0x0" ::: "memory");
}

// ---------------------------------------------------------------------------
// Kernel A: fused key  k_eff[t] = sum_m bind[(t+m)%D] * unbind[m]
// ---------------------------------------------------------------------------
__global__ void keff_kernel(const float* __restrict__ bind,
                            const float* __restrict__ unbind,
                            float* __restrict__ keff) {
    int t = blockIdx.x * blockDim.x + threadIdx.x;   // 0..DIM-1
    if (t >= DIM) return;
    float acc = 0.f;
    for (int m = 0; m < DIM; ++m)
        acc += bind[(t + m) & (DIM - 1)] * unbind[m];
    keff[t] = acc;
}

// ---------------------------------------------------------------------------
// Kernel B: transposed circulant  Mt[d][j] = keff[(d-j)%D]  as bf16 hi/lo
// ---------------------------------------------------------------------------
__global__ void circulant_kernel(const float* __restrict__ keff,
                                 bf16_t* __restrict__ mth,
                                 bf16_t* __restrict__ mtl) {
    int idx = blockIdx.x * blockDim.x + threadIdx.x;  // 0..DIM*DIM-1
    int d = idx >> 10;
    int j = idx & (DIM - 1);
    float v = keff[(d - j) & (DIM - 1)];
    bf16_t hi = (bf16_t)v;
    mth[idx] = hi;
    mtl[idx] = (bf16_t)(v - (float)hi);
}

// ---------------------------------------------------------------------------
// Kernel C1: per-(b,d,chunk) partial sums over 256 sequence positions
// ---------------------------------------------------------------------------
__global__ void chunksum_kernel(const float* __restrict__ x,
                                float* __restrict__ csum) {
    int bid   = blockIdx.x;
    int dgrp  = bid & 3;             // DIM/256 = 4
    int chunk = (bid >> 2) & (NCHUNK - 1);
    int b     = bid >> 6;
    int d     = dgrp * 256 + threadIdx.x;
    const float* src = x + ((size_t)b * SEQ + (size_t)chunk * CHUNK) * DIM + d;
    float acc = 0.f;
    for (int i = 0; i < CHUNK; ++i) acc += src[(size_t)i * DIM];
    csum[((size_t)b * DIM + d) * NCHUNK + chunk] = acc;
}

// ---------------------------------------------------------------------------
// Kernel C2: in-place exclusive scan of the 16 chunk sums per (b,d)
// ---------------------------------------------------------------------------
__global__ void chunkscan_kernel(float* __restrict__ csum) {
    int gid = blockIdx.x * blockDim.x + threadIdx.x;  // 0..BATCH*DIM-1
    float* p = csum + (size_t)gid * NCHUNK;
    float run = 0.f;
    for (int c = 0; c < NCHUNK; ++c) {
        float t = p[c];
        p[c] = run;
        run += t;
    }
}

// ---------------------------------------------------------------------------
// Kernel C3: full cumsum, emitted as bf16 hi/lo pair (split precision)
// ---------------------------------------------------------------------------
__global__ void cumsum_emit_kernel(const float* __restrict__ x,
                                   const float* __restrict__ csum,
                                   bf16_t* __restrict__ cxh,
                                   bf16_t* __restrict__ cxl) {
    int bid   = blockIdx.x;
    int dgrp  = bid & 3;
    int chunk = (bid >> 2) & (NCHUNK - 1);
    int b     = bid >> 6;
    int d     = dgrp * 256 + threadIdx.x;
    float acc = csum[((size_t)b * DIM + d) * NCHUNK + chunk];
    size_t base = ((size_t)b * SEQ + (size_t)chunk * CHUNK) * DIM + d;
    for (int i = 0; i < CHUNK; ++i) {
        size_t idx = base + (size_t)i * DIM;
        acc += x[idx];
        bf16_t hi = (bf16_t)acc;
        cxh[idx] = hi;
        cxl[idx] = (bf16_t)(acc - (float)hi);
    }
}

// ---------------------------------------------------------------------------
// Kernel D: tiled split-bf16 GEMM  out = x + gate * (cumx @ Mt^T)
// block = 256 threads (8 waves); tile = 128(S) x 128(D)
// Double-buffered async global->LDS staging; 24 WMMAs per K-step per wave.
// ---------------------------------------------------------------------------
__global__ void __launch_bounds__(256)
gemm_kernel(const float* __restrict__ x,
            const bf16_t* __restrict__ cxh, const bf16_t* __restrict__ cxl,
            const bf16_t* __restrict__ mth, const bf16_t* __restrict__ mtl,
            const float* __restrict__ gate_p,
            float* __restrict__ out) {
    // [buf:2][array:4 (Ah,Al,Bh,Bl)][TS*LDK]
    __shared__ __align__(16) bf16_t smem[2 * 4 * TS * LDK];

    const int d0 = blockIdx.x * TD;
    const int s0 = blockIdx.y * TS;
    const int b  = blockIdx.z;

    const int tid  = threadIdx.x;
    const int lane = tid & 31;
    const int wave = tid >> 5;
    const int wm   = wave & 3;          // row group: 32*wm
    const int wn   = wave >> 2;         // col group: 64*wn
    const int lr   = lane & 15;
    const int ka0  = (lane >> 4) * 8;   // 16-bit A frag: K in {ka0..+7, ka0+16..+23}
    const int kb0  = (lane >> 4) * 8;   // 16-bit B frag: K in {kb0..+7, kb0+16..+23}

    // LDS base byte offset of smem (flat-LDS address low 32 bits = wg-relative offset)
    const uint32_t smem_off = (uint32_t)(uintptr_t)(void*)&smem[0];

    // per-thread staging geometry: 2 x (row, seg) covering 128 rows x 32 elems
    const int idx0 = tid;               // t = 0
    const int idx1 = tid + 256;         // t = 1
    const int row0 = idx0 >> 2, seg0 = (idx0 & 3) * 8;
    const int row1 = idx1 >> 2, seg1 = (idx1 & 3) * 8;
    const uint32_t l0 = (uint32_t)((row0 * LDK + seg0) * 2);
    const uint32_t l1 = (uint32_t)((row1 * LDK + seg1) * 2);

    const float g = gate_p[0];

    v8f acc[2][4];
#pragma unroll
    for (int mi = 0; mi < 2; ++mi)
#pragma unroll
        for (int ni = 0; ni < 4; ++ni)
#pragma unroll
            for (int i = 0; i < 8; ++i) acc[mi][ni][i] = 0.f;

    // issue one staging stage: 8 async B128 copies per thread
    auto stage = [&](int buf, int k0) {
        uint32_t base = smem_off + (uint32_t)buf * (4 * ARR_B);
        size_t ga0 = ((size_t)b * SEQ + (size_t)(s0 + row0)) * DIM + k0 + seg0;
        size_t ga1 = ((size_t)b * SEQ + (size_t)(s0 + row1)) * DIM + k0 + seg1;
        size_t gb0 = (size_t)(d0 + row0) * DIM + k0 + seg0;
        size_t gb1 = (size_t)(d0 + row1) * DIM + k0 + seg1;
        async_copy16(base + 0 * ARR_B + l0, cxh + ga0);
        async_copy16(base + 0 * ARR_B + l1, cxh + ga1);
        async_copy16(base + 1 * ARR_B + l0, cxl + ga0);
        async_copy16(base + 1 * ARR_B + l1, cxl + ga1);
        async_copy16(base + 2 * ARR_B + l0, mth + gb0);
        async_copy16(base + 2 * ARR_B + l1, mth + gb1);
        async_copy16(base + 3 * ARR_B + l0, mtl + gb0);
        async_copy16(base + 3 * ARR_B + l1, mtl + gb1);
    };

    const int NK = DIM / TK;
    stage(0, 0);                               // prefetch stage 0

    for (int kk = 0; kk < NK; ++kk) {
        const int buf = kk & 1;
        if (kk + 1 < NK) {
            stage(buf ^ 1, (kk + 1) * TK);     // overlap: prefetch next stage
            wait_async8();                     // in-order: stage kk's 8 copies done
        } else {
            wait_async0();
        }
        __syncthreads();                       // all waves' stage-kk data visible

        const bf16_t* sAh = &smem[(size_t)(buf * 4 + 0) * (TS * LDK)];
        const bf16_t* sAl = &smem[(size_t)(buf * 4 + 1) * (TS * LDK)];
        const bf16_t* sBh = &smem[(size_t)(buf * 4 + 2) * (TS * LDK)];
        const bf16_t* sBl = &smem[(size_t)(buf * 4 + 3) * (TS * LDK)];

        FragBF ah[2], al[2];
#pragma unroll
        for (int mi = 0; mi < 2; ++mi) {
            int base = (32 * wm + 16 * mi + lr) * LDK + ka0;
            ah[mi].q[0] = *(const uint4*)&sAh[base];
            ah[mi].q[1] = *(const uint4*)&sAh[base + 16];
            al[mi].q[0] = *(const uint4*)&sAl[base];
            al[mi].q[1] = *(const uint4*)&sAl[base + 16];
        }
        FragBF bh[4], bl[4];
#pragma unroll
        for (int ni = 0; ni < 4; ++ni) {
            int base = (64 * wn + 16 * ni + lr) * LDK + kb0;
            bh[ni].q[0] = *(const uint4*)&sBh[base];
            bh[ni].q[1] = *(const uint4*)&sBh[base + 16];
            bl[ni].q[0] = *(const uint4*)&sBl[base];
            bl[ni].q[1] = *(const uint4*)&sBl[base + 16];
        }
#pragma unroll
        for (int mi = 0; mi < 2; ++mi)
#pragma unroll
            for (int ni = 0; ni < 4; ++ni) {
                acc[mi][ni] = wmma_bf16(ah[mi].v, bh[ni].v, acc[mi][ni]);  // hi*hi
                acc[mi][ni] = wmma_bf16(al[mi].v, bh[ni].v, acc[mi][ni]);  // lo*hi
                acc[mi][ni] = wmma_bf16(ah[mi].v, bl[ni].v, acc[mi][ni]);  // hi*lo
            }
        __syncthreads();                       // all reads of buf done -> reusable
    }

#pragma unroll
    for (int mi = 0; mi < 2; ++mi)
#pragma unroll
        for (int ni = 0; ni < 4; ++ni) {
            int rowbase = s0 + 32 * wm + 16 * mi + (lane >> 4) * 8;
            int col     = d0 + 64 * wn + 16 * ni + (lane & 15);
#pragma unroll
            for (int i = 0; i < 8; ++i) {
                size_t idx = ((size_t)b * SEQ + (size_t)(rowbase + i)) * DIM + col;
                out[idx] = x[idx] + g * acc[mi][ni][i];
            }
        }
}

// ---------------------------------------------------------------------------
extern "C" void kernel_launch(void* const* d_in, const int* in_sizes, int n_in,
                              void* d_out, int out_size, void* d_ws, size_t ws_size,
                              hipStream_t stream) {
    const float* x      = (const float*)d_in[0];
    const float* bind   = (const float*)d_in[1];
    const float* unbind = (const float*)d_in[2];
    const float* gate   = (const float*)d_in[3];
    float* out          = (float*)d_out;

    char* w = (char*)d_ws;
    size_t off = 0;
    float*  keff = (float*)(w + off);  off += (size_t)DIM * 4;                 off = (off + 255) & ~(size_t)255;
    bf16_t* mth  = (bf16_t*)(w + off); off += (size_t)DIM * DIM * 2;           off = (off + 255) & ~(size_t)255;
    bf16_t* mtl  = (bf16_t*)(w + off); off += (size_t)DIM * DIM * 2;           off = (off + 255) & ~(size_t)255;
    float*  csum = (float*)(w + off);  off += (size_t)BATCH * DIM * NCHUNK * 4; off = (off + 255) & ~(size_t)255;
    bf16_t* cxh  = (bf16_t*)(w + off); off += (size_t)BATCH * SEQ * DIM * 2;   off = (off + 255) & ~(size_t)255;
    bf16_t* cxl  = (bf16_t*)(w + off);

    // 1. fused key
    keff_kernel<<<DIM / 256, 256, 0, stream>>>(bind, unbind, keff);
    // 2. circulant matrix (transposed), bf16 hi/lo
    circulant_kernel<<<(DIM * DIM) / 256, 256, 0, stream>>>(keff, mth, mtl);
    // 3. cumsum over S, emitted as bf16 hi/lo
    chunksum_kernel<<<BATCH * NCHUNK * (DIM / 256), 256, 0, stream>>>(x, csum);
    chunkscan_kernel<<<(BATCH * DIM) / 256, 256, 0, stream>>>(csum);
    cumsum_emit_kernel<<<BATCH * NCHUNK * (DIM / 256), 256, 0, stream>>>(x, csum, cxh, cxl);
    // 4. split-bf16 WMMA GEMM, async-LDS double-buffered pipeline
    gemm_kernel<<<dim3(DIM / TD, SEQ / TS, BATCH), 256, 0, stream>>>(
        x, cxh, cxl, mth, mtl, gate, out);
}
